// VectorQuantizer_40707700031949
// MI455X (gfx1250) — matrix operand order repaired and checked by reading, not compiled, not run on hardware
//
#include <hip/hip_runtime.h>

typedef __attribute__((ext_vector_type(16))) _Float16 v16h;
typedef __attribute__((ext_vector_type(8)))  _Float16 v8h;
typedef __attribute__((ext_vector_type(8)))  float    v8f;

#define K_CODES 512
#define D_DIM   64
#define N_ROWS  262144
#define CB_STRIDE 72            // halves per codebook row in LDS (144B: 16B aligned, bank-spread)
#define BLOCKS 512
#define WAVES_PER_BLOCK 8
#define TILES_PER_WAVE 4        // 512 * 8 * 4 * 16 = 262144 rows

__global__ __launch_bounds__(256) void vq_main_kernel(
    const float* __restrict__ x, const float* __restrict__ cb,
    float* __restrict__ out, float* __restrict__ partial)
{
    __shared__ _Float16 lds_cb[K_CODES * CB_STRIDE];
    __shared__ float    lds_esq[K_CODES];
    __shared__ float    lds_wloss[WAVES_PER_BLOCK];

    const int tid = threadIdx.x;

    // Stage codebook as f16 (WMMA B operand source) + fp32 ||e||^2.
    for (int i = tid; i < K_CODES * D_DIM; i += 256) {
        int r = i >> 6, c = i & 63;
        lds_cb[r * CB_STRIDE + c] = (_Float16)cb[i];
    }
    for (int r = tid; r < K_CODES; r += 256) {
        float s = 0.f;
        #pragma unroll
        for (int c = 0; c < D_DIM; ++c) { float v = cb[r * D_DIM + c]; s += v * v; }
        lds_esq[r] = s;
    }
    __syncthreads();

    const int lane = tid & 31;
    const int wave = tid >> 5;
    const int n = lane & 15;    // N-column / M-row selector within 16-wide tile
    const int g = lane >> 4;    // half-wave group
    const int kb = g * 8;
    float lossAcc = 0.f;

    for (int t = 0; t < TILES_PER_WAVE; ++t) {
        const int tileIdx = (blockIdx.x * WAVES_PER_BLOCK + wave) * TILES_PER_WAVE + t;
        const int rowBase = tileIdx * 16;

        // ---- Build A (16x32 f16, two K-chunks covering D=64) per CDNA5 layout:
        // lanes 0-15: row M=n, K = {kb..kb+7, 16+kb..16+kb+7}, kb = g*8
        const float* xr = x + (size_t)(rowBase + n) * D_DIM;
        // Pull next tile's A stream toward the WGP while this tile computes.
        if (t + 1 < TILES_PER_WAVE)
            __builtin_prefetch(xr + 16 * D_DIM, 0, 3);

        v16h a0, a1;
        float sumX2 = 0.f;
        #pragma unroll
        for (int q = 0; q < 2; ++q) {
            const float4* p0 = (const float4*)(xr + q * 32 + kb);
            const float4* p1 = (const float4*)(xr + q * 32 + 16 + kb);
            float4 f0 = p0[0], f1 = p0[1], f2 = p1[0], f3 = p1[1];
            float fl[16] = { f0.x,f0.y,f0.z,f0.w, f1.x,f1.y,f1.z,f1.w,
                             f2.x,f2.y,f2.z,f2.w, f3.x,f3.y,f3.z,f3.w };
            v16h& A = q ? a1 : a0;
            #pragma unroll
            for (int h = 0; h < 16; ++h) { A[h] = (_Float16)fl[h]; sumX2 += fl[h] * fl[h]; }
        }
        // lane m and m+16 hold disjoint halves of row m
        sumX2 += __shfl_xor(sumX2, 16);

        // ---- Scan 32 code tiles of 16 codes each with v_wmma_f32_16x16x32_f16.
        // Unroll 2: two independent WMMA accumulator chains interleave, filling
        // the WMMA->VALU hazard slots with the sibling tile's loads/compares.
        float bestScore[8];
        int   bestIdx[8];
        #pragma unroll
        for (int v = 0; v < 8; ++v) { bestScore[v] = 3.4e38f; bestIdx[v] = 0; }

        #pragma unroll 2
        for (int ct = 0; ct < 32; ++ct) {
            const int code = ct * 16 + n;
            // B layout: lane holds column (ct*16+n), K rows {g*8..g*8+7, 16+g*8..16+g*8+7} (+32 for chunk 1)
            const _Float16* bp = &lds_cb[code * CB_STRIDE + g * 8];
            v8h l0 = *(const v8h*)(bp);      v8h h0 = *(const v8h*)(bp + 16);
            v8h l1 = *(const v8h*)(bp + 32); v8h h1 = *(const v8h*)(bp + 48);
            v16h b0 = __builtin_shufflevector(l0, h0, 0,1,2,3,4,5,6,7,8,9,10,11,12,13,14,15);
            v16h b1 = __builtin_shufflevector(l1, h1, 0,1,2,3,4,5,6,7,8,9,10,11,12,13,14,15);

            v8f c = {};
            c = __builtin_amdgcn_wmma_f32_16x16x32_f16(false, a0, false, b0, (short)0, c, false, false);
            c = __builtin_amdgcn_wmma_f32_16x16x32_f16(false, a1, false, b1, (short)0, c, false, false);

            const float esq = lds_esq[code];
            #pragma unroll
            for (int v = 0; v < 8; ++v) {
                float score = __builtin_fmaf(-2.f, c[v], esq); // ||e||^2 - 2 x.e
                bool take = score < bestScore[v];              // codes increase: '<' keeps first min
                bestScore[v] = take ? score : bestScore[v];
                bestIdx[v]   = take ? code  : bestIdx[v];
            }
        }

        // ---- Branchless argmin across the 16 columns (lanes within each group)
        #pragma unroll
        for (int v = 0; v < 8; ++v) {
            float s = bestScore[v]; int id = bestIdx[v];
            #pragma unroll
            for (int mask = 1; mask <= 8; mask <<= 1) {
                float os = __shfl_xor(s, mask);
                int   oi = __shfl_xor(id, mask);
                bool take = (os < s) | ((os == s) & (oi < id));  // bitwise: no branches
                s  = take ? os : s;
                id = take ? oi : id;
            }
            bestScore[v] = s; bestIdx[v] = id;
        }

        // ---- Outputs: group g owns rows g*8+v. 16 lanes x float4 = one 64-wide row.
        #pragma unroll
        for (int v = 0; v < 8; ++v) {
            const int rowLocal = g * 8 + v;
            const size_t row = (size_t)rowBase + rowLocal;
            const int idx = bestIdx[v];

            float4 qv = *(const float4*)(cb + (size_t)idx * D_DIM + n * 4);
            *(float4*)(out + 1 + row * D_DIM + n * 4) = qv;

            if (n == 0)
                out[1 + (size_t)N_ROWS * D_DIM + row] = (float)idx;

            // loss_row = ||x||^2 + (||e||^2 - 2 x.e) ; sumX2 for row r sits in lane (r&15)
            float sx = __shfl(sumX2, rowLocal & 15);
            lossAcc += (n == v) ? (sx + bestScore[v]) : 0.f;
        }
    }

    // ---- Deterministic block partial for the loss
    #pragma unroll
    for (int mask = 16; mask >= 1; mask >>= 1) lossAcc += __shfl_xor(lossAcc, mask);
    if (lane == 0) lds_wloss[wave] = lossAcc;
    __syncthreads();
    if (tid == 0) {
        float s = 0.f;
        #pragma unroll
        for (int w = 0; w < WAVES_PER_BLOCK; ++w) s += lds_wloss[w];
        partial[blockIdx.x] = s;
    }
}

__global__ __launch_bounds__(256) void vq_reduce_kernel(
    const float* __restrict__ partial, float* __restrict__ out)
{
    __shared__ float s[256];
    const int tid = threadIdx.x;
    s[tid] = partial[tid] + partial[tid + 256];
    __syncthreads();
    for (int o = 128; o > 0; o >>= 1) {
        if (tid < o) s[tid] += s[tid + o];
        __syncthreads();
    }
    if (tid == 0)
        out[0] = s[0] * (1.25f / ((float)N_ROWS * (float)D_DIM));
}

extern "C" void kernel_launch(void* const* d_in, const int* in_sizes, int n_in,
                              void* d_out, int out_size, void* d_ws, size_t ws_size,
                              hipStream_t stream) {
    const float* x  = (const float*)d_in[0];   // [N, 64]
    const float* cb = (const float*)d_in[1];   // [512, 64]
    float* out = (float*)d_out;                // [loss | quantized N*64 | indices N]
    float* partial = (float*)d_ws;             // BLOCKS floats

    vq_main_kernel<<<BLOCKS, 256, 0, stream>>>(x, cb, out, partial);
    vq_reduce_kernel<<<1, 256, 0, stream>>>(partial, out);
}
